// SimpleGCNESAModel_15522011807839
// MI455X (gfx1250) — compile-verified, matrix operand order checked by reading, not run on hardware
//
#include <hip/hip_runtime.h>

// ---------------------------------------------------------------------------
// SimpleGCNESAModel forward for MI455X (gfx1250, wave32, WMMA).
//
// Dense GEMMs (GCN x@W^T, QKV proj, out proj) run on v_wmma_f32_16x16x32_f16.
// Block = 8 waves covering a 16(M) x 128(N) C-strip; the 16x128 A tile (4 KB)
// is staged in LDS once per block via global_load_async_to_lds_b128,
// each wave reads its W rows from global (weights are L2-resident).
// Irregular parts (edge scatter-add, softmax over 500 keys, LN+MLP head)
// are VALU/LDS/f32-atomic kernels.
// ---------------------------------------------------------------------------

#define N_ATOMS   64000
#define NBLK      8000
#define HDIM      128
#define H3        (3 * HDIM)
#define BGRAPH    16
#define SBLK      500
#define NHEADS    8
#define DHEAD     16
#define APB       8          // atoms per block

typedef __attribute__((ext_vector_type(16))) _Float16 v16h;
typedef __attribute__((ext_vector_type(8)))  float    v8f;
typedef __attribute__((ext_vector_type(4)))  float    f32x4;
typedef __attribute__((ext_vector_type(4)))  int      i32x4;

#define AS1 __attribute__((address_space(1)))
#define AS3 __attribute__((address_space(3)))

#if defined(__has_builtin)
#  if __has_builtin(__builtin_amdgcn_global_load_async_to_lds_b128)
#    define HAVE_ASYNC_LDS 1
#  else
#    define HAVE_ASYNC_LDS 0
#  endif
#  if __has_builtin(__builtin_amdgcn_s_wait_asynccnt)
#    define HAVE_WAIT_ASYNC 1
#  else
#    define HAVE_WAIT_ASYNC 0
#  endif
#else
#  define HAVE_ASYNC_LDS 0
#  define HAVE_WAIT_ASYNC 0
#endif

// ---------------------------------------------------------------------------
// WMMA GEMM: C[M,N] = act( A[M,128] @ W[N,128]^T + bias ), A/W f16 row-major.
// grid = (M/16, N/128), block = 256 threads (8 waves).
// Wave w computes the 16x16 tile at columns n0 = blockIdx.y*128 + w*16.
// Per-lane fragment packing per CDNA5 ISA 7.12.2 (16-bit A 16x32):
//   lanes 0-15 : row M=lane,    K = {k0..k0+7, k0+16..k0+23}
//   lanes 16-31: row M=lane-16, K = {k0+8..k0+15, k0+24..k0+31}
// B (32x16 KxN) mirrors this with the lane index selecting the N column,
// i.e. the same packing applied to rows of W. C: VGPR r -> M = r + 8*half.
// ---------------------------------------------------------------------------
__global__ void gemm_wmma_kernel(const _Float16* __restrict__ A,
                                 const _Float16* __restrict__ W,
                                 const float* __restrict__ bias,
                                 float* __restrict__ C,
                                 _Float16* __restrict__ Ch,
                                 int M, int N, int relu)
{
    __shared__ _Float16 lds_a[16 * HDIM];   // 4 KB A tile, shared by 8 waves

    const int tid  = threadIdx.x;
    const int wave = tid >> 5;
    const int lane = tid & 31;
    const int half = lane >> 4;      // 0 or 1
    const int l16  = lane & 15;
    const int m0 = blockIdx.x * 16;
    const int n0 = blockIdx.y * 128 + wave * 16;

    // ---- stage A tile: 16 rows x 128 halfs = 4 KB contiguous; 16 B/thread ----
    {
        const _Float16* src = A + (size_t)m0 * HDIM + (size_t)tid * 8;
        _Float16*       dst = &lds_a[tid * 8];
#if HAVE_ASYNC_LDS
        // async DMA into LDS (ASYNCcnt-tracked): (global src, lds dst, off, cpol)
        __builtin_amdgcn_global_load_async_to_lds_b128(
            (AS1 i32x4*)(void*)src,
            (AS3 i32x4*)(void*)dst, 0, 0);
#if HAVE_WAIT_ASYNC
        __builtin_amdgcn_s_wait_asynccnt(0);
#else
        asm volatile("s_wait_asynccnt 0" ::: "memory");
#endif
#else
        *(f32x4*)dst = *(const f32x4*)src;
#endif
    }
    __syncthreads();

    const _Float16* __restrict__ wrow = W + (size_t)(n0 + l16) * HDIM;
    const _Float16* __restrict__ arow = &lds_a[l16 * HDIM];

    v8f c = {};
#pragma unroll
    for (int k0 = 0; k0 < HDIM; k0 += 32) {
        // gfx1250 prefetch path (global_prefetch_b8) for the next W chunk
        __builtin_prefetch(wrow + k0 + 32, 0, 0);
        v16h a, b;
        const int base = k0 + half * 8;
#pragma unroll
        for (int j = 0; j < 8; ++j) {
            a[j]     = arow[base + j];
            a[j + 8] = arow[base + 16 + j];
            b[j]     = wrow[base + j];
            b[j + 8] = wrow[base + 16 + j];
        }
        // (neg_a, A, neg_b, B, c_mod, C, reuse_a, reuse_b)
        c = __builtin_amdgcn_wmma_f32_16x16x32_f16(false, a, false, b,
                                                   (short)0, c, false, false);
    }

#pragma unroll
    for (int r = 0; r < 8; ++r) {
        const int row = m0 + half * 8 + r;
        const int col = n0 + l16;
        float v = c[r];
        if (bias) v += bias[col];
        if (relu) v = v > 0.0f ? v : 0.0f;
        if (C)  C[(size_t)row * N + col] = v;
        if (Ch) Ch[(size_t)row * N + col] = (_Float16)v;
    }
    (void)M;
}

// --------------------------- elementwise kernels ---------------------------

__global__ void convert_f16_kernel(const float* __restrict__ src,
                                   _Float16* __restrict__ dst, int n)
{
    int i = blockIdx.x * blockDim.x + threadIdx.x;
    if (i < n) dst[i] = (_Float16)src[i];
}

__global__ void embed_kernel(const int* __restrict__ A, const int* __restrict__ pos,
                             const int* __restrict__ B,
                             const float* __restrict__ atom_emb,
                             const float* __restrict__ pos_emb,
                             const float* __restrict__ blk_emb,
                             _Float16* __restrict__ xh)
{
    int idx = blockIdx.x * blockDim.x + threadIdx.x;
    if (idx >= N_ATOMS * HDIM) return;
    int i = idx / HDIM, h = idx - i * HDIM;
    float v = atom_emb[A[i] * HDIM + h]
            + pos_emb[pos[i] * HDIM + h]
            + blk_emb[B[i >> 3] * HDIM + h];   // block_id = atom/8
    xh[idx] = (_Float16)v;
}

__global__ void deg_init_kernel(float* __restrict__ deg)
{
    int i = blockIdx.x * blockDim.x + threadIdx.x;
    if (i < N_ATOMS) deg[i] = 1.0f;            // self loop
}

__global__ void deg_edges_kernel(const int* __restrict__ dst,
                                 float* __restrict__ deg, int E)
{
    int e = blockIdx.x * blockDim.x + threadIdx.x;
    if (e < E) atomicAdd(&deg[dst[e]], 1.0f);
}

__global__ void dinv_kernel(const float* __restrict__ deg, float* __restrict__ dinv)
{
    int i = blockIdx.x * blockDim.x + threadIdx.x;
    if (i < N_ATOMS) dinv[i] = rsqrtf(deg[i]);
}

// acc[i,h] = bias[h] + y[i,h] * dinv[i]^2   (self-loop message + bias)
__global__ void gcn_self_kernel(const float* __restrict__ y,
                                const float* __restrict__ dinv,
                                const float* __restrict__ bias,
                                float* __restrict__ acc)
{
    int idx = blockIdx.x * blockDim.x + threadIdx.x;
    if (idx >= N_ATOMS * HDIM) return;
    int i = idx / HDIM, h = idx - i * HDIM;
    float di = dinv[i];
    acc[idx] = bias[h] + y[idx] * di * di;
}

// acc[dst,h] += y[src,h] * dinv[src] * dinv[dst]
__global__ void gcn_edge_kernel(const float* __restrict__ y,
                                const float* __restrict__ dinv,
                                const int* __restrict__ src,
                                const int* __restrict__ dst,
                                float* __restrict__ acc, int E)
{
    int idx = blockIdx.x * blockDim.x + threadIdx.x;
    if (idx >= E * HDIM) return;
    int e = idx / HDIM, h = idx - e * HDIM;
    int s = src[e], d = dst[e];
    atomicAdd(&acc[(size_t)d * HDIM + h],
              y[(size_t)s * HDIM + h] * dinv[s] * dinv[d]);
}

__global__ void relu_f16_kernel(const float* __restrict__ acc,
                                _Float16* __restrict__ xh, int n)
{
    int i = blockIdx.x * blockDim.x + threadIdx.x;
    if (i < n) xh[i] = (_Float16)fmaxf(acc[i], 0.0f);
}

// blk[b,h] = mean over 8 atoms of block b
__global__ void pool_kernel(const _Float16* __restrict__ xh,
                            _Float16* __restrict__ blkh)
{
    int idx = blockIdx.x * blockDim.x + threadIdx.x;
    if (idx >= NBLK * HDIM) return;
    int b = idx / HDIM, h = idx - b * HDIM;
    float s = 0.0f;
#pragma unroll
    for (int a = 0; a < APB; ++a)
        s += (float)xh[(size_t)(b * APB + a) * HDIM + h];
    blkh[idx] = (_Float16)(s * (1.0f / APB));
}

// --------------------------- attention (per graph/head) --------------------
// qkv: (NBLK, 3H) f32; writes oh (NBLK, H) f16 in (block, head*16+d) layout.
__global__ void attention_kernel(const float* __restrict__ qkv,
                                 _Float16* __restrict__ oh)
{
    __shared__ float kbuf[SBLK * DHEAD];   // 32000 B
    __shared__ float scores[SBLK];
    __shared__ float qv[DHEAD];
    __shared__ float red[256];

    const int g   = blockIdx.x / NHEADS;
    const int hh  = blockIdx.x % NHEADS;
    const int tid = threadIdx.x;
    const size_t rowbase = (size_t)g * SBLK;

    for (int t = tid; t < SBLK * DHEAD; t += 256) {
        int s = t / DHEAD, dd = t - s * DHEAD;
        kbuf[t] = qkv[(rowbase + s) * H3 + HDIM + hh * DHEAD + dd];
    }
    __syncthreads();

    for (int q = 0; q < SBLK; ++q) {
        if (tid < DHEAD)
            qv[tid] = qkv[(rowbase + q) * H3 + hh * DHEAD + tid];
        __syncthreads();

        for (int s = tid; s < SBLK; s += 256) {
            float dot = 0.0f;
#pragma unroll
            for (int dd = 0; dd < DHEAD; ++dd)
                dot += qv[dd] * kbuf[s * DHEAD + dd];
            scores[s] = dot * 0.25f;           // 1/sqrt(16)
        }
        __syncthreads();

        // max-reduce
        float m = -3.4e38f;
        for (int s = tid; s < SBLK; s += 256) m = fmaxf(m, scores[s]);
        red[tid] = m; __syncthreads();
        for (int off = 128; off > 0; off >>= 1) {
            if (tid < off) red[tid] = fmaxf(red[tid], red[tid + off]);
            __syncthreads();
        }
        const float mx = red[0];
        __syncthreads();

        // exp + sum-reduce
        float p = 0.0f;
        for (int s = tid; s < SBLK; s += 256) {
            float e = __expf(scores[s] - mx);
            scores[s] = e;
            p += e;
        }
        red[tid] = p; __syncthreads();
        for (int off = 128; off > 0; off >>= 1) {
            if (tid < off) red[tid] += red[tid + off];
            __syncthreads();
        }
        const float inv = 1.0f / red[0];
        __syncthreads();

        // o[dd] = sum_s scores[s] * V[s,dd]; 16 partial groups per dd
        const int dd  = tid & (DHEAD - 1);
        const int grp = tid >> 4;              // 0..15
        float acc = 0.0f;
        for (int s = grp; s < SBLK; s += 16)
            acc += scores[s] * qkv[(rowbase + s) * H3 + 2 * HDIM + hh * DHEAD + dd];
        red[dd * 16 + grp] = acc;
        __syncthreads();
        if (tid < DHEAD) {
            float o = 0.0f;
#pragma unroll
            for (int j = 0; j < 16; ++j) o += red[tid * 16 + j];
            oh[(rowbase + q) * HDIM + hh * DHEAD + tid] = (_Float16)(o * inv);
        }
        __syncthreads();
    }
}

// g[graph,h] = mean over S blocks of attout
__global__ void graph_mean_kernel(const float* __restrict__ attout,
                                  float* __restrict__ gbuf)
{
    int g = blockIdx.x, h = threadIdx.x;
    float s = 0.0f;
    for (int b = 0; b < SBLK; ++b)
        s += attout[((size_t)g * SBLK + b) * HDIM + h];
    gbuf[g * HDIM + h] = s * (1.0f / SBLK);
}

// LayerNorm + 3-layer MLP head; one block (128 threads) per graph.
__global__ void head_kernel(const float* __restrict__ gbuf,
                            const float* __restrict__ ln_g, const float* __restrict__ ln_b,
                            const float* __restrict__ w1, const float* __restrict__ b1,
                            const float* __restrict__ w2, const float* __restrict__ b2,
                            const float* __restrict__ w3, const float* __restrict__ b3,
                            float* __restrict__ out)
{
    __shared__ float sg[HDIM];
    __shared__ float sh1[HDIM];
    __shared__ float red[HDIM];
    const int g = blockIdx.x, t = threadIdx.x;

    float v = gbuf[g * HDIM + t];
    red[t] = v; __syncthreads();
    for (int off = 64; off > 0; off >>= 1) {
        if (t < off) red[t] += red[t + off];
        __syncthreads();
    }
    const float mu = red[0] * (1.0f / HDIM);
    __syncthreads();
    const float dv = v - mu;
    red[t] = dv * dv; __syncthreads();
    for (int off = 64; off > 0; off >>= 1) {
        if (t < off) red[t] += red[t + off];
        __syncthreads();
    }
    const float var = red[0] * (1.0f / HDIM);
    __syncthreads();

    sg[t] = dv * rsqrtf(var + 1e-5f) * ln_g[t] + ln_b[t];
    __syncthreads();

    float d1 = 0.0f;
    for (int k = 0; k < HDIM; ++k) d1 += sg[k] * w1[t * HDIM + k];
    sh1[t] = fmaxf(d1 + b1[t], 0.0f);
    __syncthreads();

    float d2 = 0.0f;
    for (int k = 0; k < HDIM; ++k) d2 += sh1[k] * w2[t * HDIM + k];
    d2 = fmaxf(d2 + b2[t], 0.0f);

    red[t] = d2 * w3[t]; __syncthreads();
    for (int off = 64; off > 0; off >>= 1) {
        if (t < off) red[t] += red[t + off];
        __syncthreads();
    }
    if (t == 0) out[g] = red[0] + b3[0];
}

// ---------------------------------------------------------------------------

extern "C" void kernel_launch(void* const* d_in, const int* in_sizes, int n_in,
                              void* d_out, int out_size, void* d_ws, size_t ws_size,
                              hipStream_t stream)
{
    const int*   A        = (const int*)d_in[1];
    const int*   apos     = (const int*)d_in[2];
    const int*   B        = (const int*)d_in[3];
    const int*   esrc     = (const int*)d_in[7];
    const int*   edst     = (const int*)d_in[8];
    const float* atom_emb = (const float*)d_in[9];
    const float* pos_emb  = (const float*)d_in[10];
    const float* blk_emb  = (const float*)d_in[11];
    const float* gcn_w1   = (const float*)d_in[12];
    const float* gcn_b1   = (const float*)d_in[13];
    const float* gcn_w2   = (const float*)d_in[14];
    const float* gcn_b2   = (const float*)d_in[15];
    const float* in_proj_w= (const float*)d_in[16];
    const float* in_proj_b= (const float*)d_in[17];
    const float* out_w    = (const float*)d_in[18];
    const float* out_b    = (const float*)d_in[19];
    const float* ln_g     = (const float*)d_in[20];
    const float* ln_b     = (const float*)d_in[21];
    const float* mlp_w1   = (const float*)d_in[22];
    const float* mlp_b1   = (const float*)d_in[23];
    const float* mlp_w2   = (const float*)d_in[24];
    const float* mlp_b2   = (const float*)d_in[25];
    const float* mlp_w3   = (const float*)d_in[26];
    const float* mlp_b3   = (const float*)d_in[27];
    const int E = in_sizes[7];
    float* out = (float*)d_out;

    // ---- workspace layout (~103 MB) ----
    char* ws = (char*)d_ws;
    auto carve = [&](size_t bytes) {
        void* p = ws;
        ws += (bytes + 255) & ~(size_t)255;
        return p;
    };
    _Float16* xh     = (_Float16*)carve((size_t)N_ATOMS * HDIM * 2);
    float*    ybuf   = (float*)   carve((size_t)N_ATOMS * HDIM * 4);
    float*    acc    = (float*)   carve((size_t)N_ATOMS * HDIM * 4);
    float*    deg    = (float*)   carve((size_t)N_ATOMS * 4);
    float*    dinv   = (float*)   carve((size_t)N_ATOMS * 4);
    _Float16* blkh   = (_Float16*)carve((size_t)NBLK * HDIM * 2);
    float*    qkv    = (float*)   carve((size_t)NBLK * H3 * 4);
    _Float16* ohbuf  = (_Float16*)carve((size_t)NBLK * HDIM * 2);
    float*    attout = (float*)   carve((size_t)NBLK * HDIM * 4);
    float*    gbuf   = (float*)   carve((size_t)BGRAPH * HDIM * 4);
    _Float16* w1h    = (_Float16*)carve((size_t)HDIM * HDIM * 2);
    _Float16* w2h    = (_Float16*)carve((size_t)HDIM * HDIM * 2);
    _Float16* iph    = (_Float16*)carve((size_t)H3 * HDIM * 2);
    _Float16* owh    = (_Float16*)carve((size_t)HDIM * HDIM * 2);

    const int T = 256;
    auto blocks = [](int n, int t) { return (n + t - 1) / t; };

    // weight conversions f32 -> f16
    convert_f16_kernel<<<blocks(HDIM * HDIM, T), T, 0, stream>>>(gcn_w1, w1h, HDIM * HDIM);
    convert_f16_kernel<<<blocks(HDIM * HDIM, T), T, 0, stream>>>(gcn_w2, w2h, HDIM * HDIM);
    convert_f16_kernel<<<blocks(H3 * HDIM, T), T, 0, stream>>>(in_proj_w, iph, H3 * HDIM);
    convert_f16_kernel<<<blocks(HDIM * HDIM, T), T, 0, stream>>>(out_w, owh, HDIM * HDIM);

    // embeddings -> xh
    embed_kernel<<<blocks(N_ATOMS * HDIM, T), T, 0, stream>>>(
        A, apos, B, atom_emb, pos_emb, blk_emb, xh);

    // degrees (self loop + in-edges), dinv = rsqrt(deg)
    deg_init_kernel<<<blocks(N_ATOMS, T), T, 0, stream>>>(deg);
    if (E > 0)
        deg_edges_kernel<<<blocks(E, T), T, 0, stream>>>(edst, deg, E);
    dinv_kernel<<<blocks(N_ATOMS, T), T, 0, stream>>>(deg, dinv);

    // ---- GCN layer 1 ----
    {
        dim3 grid(N_ATOMS / 16, HDIM / 128);
        gemm_wmma_kernel<<<grid, 256, 0, stream>>>(xh, w1h, nullptr, ybuf, nullptr,
                                                   N_ATOMS, HDIM, 0);
    }
    gcn_self_kernel<<<blocks(N_ATOMS * HDIM, T), T, 0, stream>>>(ybuf, dinv, gcn_b1, acc);
    if (E > 0)
        gcn_edge_kernel<<<blocks(E * HDIM, T), T, 0, stream>>>(ybuf, dinv, esrc, edst, acc, E);
    relu_f16_kernel<<<blocks(N_ATOMS * HDIM, T), T, 0, stream>>>(acc, xh, N_ATOMS * HDIM);

    // ---- GCN layer 2 ----
    {
        dim3 grid(N_ATOMS / 16, HDIM / 128);
        gemm_wmma_kernel<<<grid, 256, 0, stream>>>(xh, w2h, nullptr, ybuf, nullptr,
                                                   N_ATOMS, HDIM, 0);
    }
    gcn_self_kernel<<<blocks(N_ATOMS * HDIM, T), T, 0, stream>>>(ybuf, dinv, gcn_b2, acc);
    if (E > 0)
        gcn_edge_kernel<<<blocks(E * HDIM, T), T, 0, stream>>>(ybuf, dinv, esrc, edst, acc, E);
    relu_f16_kernel<<<blocks(N_ATOMS * HDIM, T), T, 0, stream>>>(acc, xh, N_ATOMS * HDIM);

    // ---- block mean pooling ----
    pool_kernel<<<blocks(NBLK * HDIM, T), T, 0, stream>>>(xh, blkh);

    // ---- QKV projection (8000 x 384) ----
    {
        dim3 grid(NBLK / 16, H3 / 128);
        gemm_wmma_kernel<<<grid, 256, 0, stream>>>(blkh, iph, in_proj_b, qkv, nullptr,
                                                   NBLK, H3, 0);
    }

    // ---- multi-head attention (16 graphs x 8 heads) ----
    attention_kernel<<<BGRAPH * NHEADS, 256, 0, stream>>>(qkv, ohbuf);

    // ---- output projection ----
    {
        dim3 grid(NBLK / 16, HDIM / 128);
        gemm_wmma_kernel<<<grid, 256, 0, stream>>>(ohbuf, owh, out_b, attout, nullptr,
                                                   NBLK, HDIM, 0);
    }

    // ---- graph mean + LN + MLP head ----
    graph_mean_kernel<<<BGRAPH, HDIM, 0, stream>>>(attout, gbuf);
    head_kernel<<<BGRAPH, HDIM, 0, stream>>>(gbuf, ln_g, ln_b,
                                             mlp_w1, mlp_b1, mlp_w2, mlp_b2,
                                             mlp_w3, mlp_b3, out);
    (void)n_in; (void)out_size; (void)ws_size; (void)d_in; (void)in_sizes;
}
